// GLATransformer_16896401342972
// MI455X (gfx1250) — matrix operand order repaired and checked by reading, not compiled, not run on hardware
//
#include <hip/hip_runtime.h>
#include <cstdint>

// Problem constants (match reference)
#define Bz   2
#define Sz   2048
#define Vz   32000
#define DIMz 1024
#define Lz   4
#define Hz   16
#define HDz  64
#define FFz  4096
#define CHz  16
#define Mz   (Bz*Sz)          // 4096 token rows

typedef unsigned short u16;
typedef __attribute__((ext_vector_type(16))) __bf16 v16bf;
typedef __attribute__((ext_vector_type(8)))  float  v8f;

static __device__ __forceinline__ u16 f2bf(float f) {
  union { float f; uint32_t u; } x; x.f = f;
  uint32_t r = x.u + 0x7fffu + ((x.u >> 16) & 1u);   // round-to-nearest-even
  return (u16)(r >> 16);
}

// CDNA5 async copy: 16 bytes global -> LDS per lane, tracked by ASYNCcnt.
static __device__ __forceinline__ void async_copy16(uint32_t lds_byte, const u16* g) {
  asm volatile("global_load_async_to_lds_b128 %0, %1, off"
               :: "v"(lds_byte), "v"((unsigned long long)(uintptr_t)g)
               : "memory");
}
static __device__ __forceinline__ void async_wait4() {   // current tile done, next in flight
  asm volatile("s_wait_asynccnt 0x4" ::: "memory");
}
static __device__ __forceinline__ void async_wait0() {
  asm volatile("s_wait_asynccnt 0x0" ::: "memory");
}

// ---------------------------------------------------------------- embed gather
__global__ __launch_bounds__(256) void embed_kernel(const int* __restrict__ tok,
                                                    const float* __restrict__ embed,
                                                    float* __restrict__ x) {
  int row = blockIdx.x;
  const float* src = embed + (size_t)tok[row] * DIMz;
  float* dst = x + (size_t)row * DIMz;
  for (int c = threadIdx.x; c < DIMz; c += 256) dst[c] = src[c];
}

// ---------------------------------------------------------------- weight transform
// W[K,N] f32 (row-major) -> Wt[N,K] bf16 (row-major), 32x32 LDS tile transpose.
__global__ __launch_bounds__(256) void wtrans_kernel(const float* __restrict__ W,
                                                     u16* __restrict__ Wt,
                                                     int K, int N) {
  __shared__ float tile[32][33];
  const int k0 = blockIdx.y * 32, n0 = blockIdx.x * 32;
  const int tc = threadIdx.x & 31, tr = threadIdx.x >> 5;
  for (int r = tr; r < 32; r += 8)
    tile[r][tc] = W[(size_t)(k0 + r) * N + n0 + tc];
  __syncthreads();
  // write Wt[n][k]; pack 2 consecutive k's into one u32 store
  for (int i = threadIdx.x; i < 512; i += 256) {
    int cc = i >> 4;          // 0..31 : n within tile
    int kp = i & 15;          // 0..15 : k pair
    uint32_t lo = f2bf(tile[2 * kp][cc]);
    uint32_t hi = f2bf(tile[2 * kp + 1][cc]);
    *(uint32_t*)&Wt[(size_t)(n0 + cc) * K + k0 + 2 * kp] = lo | (hi << 16);
  }
}

// ---------------------------------------------------------------- RMSNorm -> bf16
__global__ __launch_bounds__(256) void rmsnorm_kernel(const float* __restrict__ x,
                                                      const float* __restrict__ w,
                                                      u16* __restrict__ out) {
  int row = blockIdx.x;
  const float* xr = x + (size_t)row * DIMz;
  float ss = 0.f;
  for (int c = threadIdx.x; c < DIMz; c += 256) { float v = xr[c]; ss += v * v; }
  __shared__ float red[256];
  red[threadIdx.x] = ss;
  __syncthreads();
  for (int s = 128; s > 0; s >>= 1) {
    if (threadIdx.x < s) red[threadIdx.x] += red[threadIdx.x + s];
    __syncthreads();
  }
  float rs = rsqrtf(red[0] * (1.f / DIMz) + 1e-6f);
  for (int c = threadIdx.x; c < DIMz; c += 256)
    out[(size_t)row * DIMz + c] = f2bf(xr[c] * rs * w[c]);
}

// ---------------------------------------------------------------- elementwise
__global__ __launch_bounds__(256) void silumul_kernel(const float* __restrict__ u,
                                                      const float* __restrict__ v3,
                                                      u16* __restrict__ t, int n) {
  int i = blockIdx.x * 256 + threadIdx.x;
  if (i < n) {
    float a = u[i];
    float s = a / (1.f + __expf(-a));          // silu
    t[i] = f2bf(s * v3[i]);
  }
}

// ---------------------------------------------------------------- WMMA GEMM
// C[M,N] (f32) (+)= A[M,K](bf16, row-major) @ Wt[N,K](bf16, row-major = W^T)
// block tile 128x128, K-step 32 (== one v_wmma_f32_16x16x32_bf16 K)
// 8 waves: 2 wave-rows x 4 wave-cols; wave tile 64x32 -> 4x2 WMMA accumulators.
// Double-buffered LDS, async copies (ASYNCcnt) overlapped with WMMA compute:
//   per tile each wave issues exactly 4 async ops; s_wait_asynccnt 0x4 waits
//   for the current tile while the next tile stays in flight.
// MODE: 0 = store, 1 = accumulate (residual), 2 = store + bias (LM head)
union Frag { v16bf v; uint4 u[2]; };

template <int MODE>
__global__ __launch_bounds__(256) void gemm_kernel(const u16*  __restrict__ A,
                                                   const u16*  __restrict__ Bt,
                                                   float* __restrict__ C,
                                                   const float* __restrict__ bias,
                                                   int N, int K) {
  __shared__ __align__(16) u16 As[2][128 * 32];  // [buf][row][k] bf16 (64B per row)
  __shared__ __align__(16) u16 Bs[2][128 * 32];  // [buf][col][k] bf16 (64B per col)

  const int tid  = threadIdx.x;
  const int m0   = blockIdx.y * 128;
  const int n0   = blockIdx.x * 128;
  const int w    = tid >> 5, lane = tid & 31;
  const int wm   = (w >> 2) * 64;            // wave row offset (0/64)
  const int wn   = (w & 3) * 32;             // wave col offset (0/32/64/96)
  const int r    = lane & 15, hi = lane >> 4;

  const uint32_t asLds = (uint32_t)(uintptr_t)&As[0][0];
  const uint32_t bsLds = (uint32_t)(uintptr_t)&Bs[0][0];

  auto stage = [&](int buf, int k0) {
    const uint32_t aL = asLds + (uint32_t)buf * 8192;
    const uint32_t bL = bsLds + (uint32_t)buf * 8192;
    #pragma unroll
    for (int c = tid; c < 512; c += 256) {
      int row = c >> 2, seg = c & 3;
      async_copy16(aL + (uint32_t)c * 16,
                   A  + (size_t)(m0 + row) * K + k0 + seg * 8);
      async_copy16(bL + (uint32_t)c * 16,
                   Bt + (size_t)(n0 + row) * K + k0 + seg * 8);
    }
  };

  v8f acc[4][2] = {};

  const int nb = K / 32;
  stage(0, 0);                         // prologue: first tile in flight

  for (int kt = 0; kt < nb; kt++) {
    const int cur = kt & 1;
    __syncthreads();                   // all readers of buf[cur^1] finished
    if (kt + 1 < nb) {
      stage(cur ^ 1, (kt + 1) * 32);   // prefetch next tile (stays in flight)
      async_wait4();                   // current tile's 4 asyncs retired
    } else {
      async_wait0();
    }
    __syncthreads();                   // all waves' current-tile writes visible

    const u16* Asc = &As[cur][0];
    const u16* Bsc = &Bs[cur][0];

    // A fragment (ISA 16-bit A 16x32 layout): lane r row, hi selects K-halves
    Frag a[4], b[2];
    #pragma unroll
    for (int fm = 0; fm < 4; fm++) {
      int base = (wm + fm * 16 + r) * 32 + hi * 8;
      a[fm].u[0] = *(const uint4*)&Asc[base];        // K =  8*hi ..  8*hi+7
      a[fm].u[1] = *(const uint4*)&Asc[base + 16];   // K = 16+8*hi .. +7
    }
    // B fragment: lane r = column, K = 8*hi..8*hi+7 then 16+8*hi..16+8*hi+7 (same as A)
    #pragma unroll
    for (int fn = 0; fn < 2; fn++) {
      int base = (wn + fn * 16 + r) * 32 + hi * 8;
      b[fn].u[0] = *(const uint4*)&Bsc[base];
      b[fn].u[1] = *(const uint4*)&Bsc[base + 16];
    }
    #pragma unroll
    for (int fm = 0; fm < 4; fm++)
      #pragma unroll
      for (int fn = 0; fn < 2; fn++)
        acc[fm][fn] = __builtin_amdgcn_wmma_f32_16x16x32_bf16(
            false, a[fm].v, false, b[fn].v, (short)0, acc[fm][fn], false, false);
  }

  // epilogue: C/D layout — vgpr i: lanes 0-15 -> M=i, lanes 16-31 -> M=8+i, N=lane%16
  #pragma unroll
  for (int fm = 0; fm < 4; fm++)
    #pragma unroll
    for (int fn = 0; fn < 2; fn++) {
      int col = n0 + wn + fn * 16 + r;
      #pragma unroll
      for (int i = 0; i < 8; i++) {
        int row = m0 + wm + fm * 16 + hi * 8 + i;
        float val = acc[fm][fn][i];
        float* cp = C + (size_t)row * N + col;
        if (MODE == 1)      *cp += val;
        else if (MODE == 2) *cp  = val + bias[col];
        else                *cp  = val;
      }
    }
}

// ---------------------------------------------------------------- chunkwise GLA
// One block per (b,h); 64x64 state in LDS; sequential over 128 chunks of 16.
// Sigmoid gate is fused here (g input is the raw pre-activation projection).
__global__ __launch_bounds__(256) void gla_kernel(const float* __restrict__ q,
                                                  const float* __restrict__ k,
                                                  const float* __restrict__ v,
                                                  const float* __restrict__ g,
                                                  u16* __restrict__ o) {
  __shared__ float St[HDz * HDz];
  __shared__ float qb[CHz * HDz], kb[CHz * HDz], kd[CHz * HDz], vs[CHz * HDz];
  __shared__ float lgs[CHz * HDz];
  __shared__ float Amat[CHz * CHz];
  __shared__ float gtot[HDz];

  const int tid = threadIdx.x;
  const int bh  = blockIdx.x;
  const int b   = bh / Hz, h = bh % Hz;
  const size_t colbase = (size_t)h * HDz;

  for (int i = tid; i < HDz * HDz; i += 256) St[i] = 0.f;
  __syncthreads();

  for (int ch = 0; ch < Sz / CHz; ch++) {
    const size_t rowbase = (size_t)b * Sz + (size_t)ch * CHz;

    // phase 1: per-column cumulative log(sigmoid(g))
    if (tid < HDz) {
      float lg = 0.f;
      for (int t = 0; t < CHz; t++) {
        float gp = g[(rowbase + t) * DIMz + colbase + tid];
        float gv = 1.f / (1.f + __expf(-gp));   // fused sigmoid
        lg += __logf(gv);
        lgs[t * HDz + tid] = lg;
      }
      gtot[tid] = __expf(lg);
    }
    __syncthreads();

    // phase 2: decayed q/k and chunk v
    for (int idx = tid; idx < CHz * HDz; idx += 256) {
      int t = idx >> 6, d = idx & 63;
      size_t gi = (rowbase + t) * DIMz + colbase + d;
      float lg  = lgs[idx];
      float lgT = lgs[15 * HDz + d];
      float qv = q[gi], kv = k[gi];
      qb[idx] = qv * 0.125f * __expf(lg);     // includes d^-0.5 = 1/8
      kb[idx] = kv * __expf(-lg);
      kd[idx] = kv * __expf(lgT - lg);
      vs[idx] = v[gi];
    }
    __syncthreads();

    // phase 3: masked intra-chunk attention A = tril(qb kb^T)
    {
      int t = tid >> 4, s = tid & 15;
      float a = 0.f;
      if (s <= t)
        for (int d = 0; d < HDz; d++) a += qb[t * HDz + d] * kb[s * HDz + d];
      Amat[tid] = a;
    }
    __syncthreads();

    // phase 4: o = A @ V (intra) + qb @ S (inter), write bf16
    for (int idx = tid; idx < CHz * HDz; idx += 256) {
      int t = idx >> 6, e = idx & 63;
      float acc = 0.f;
      for (int s = 0; s <= t; s++) acc += Amat[t * 16 + s] * vs[s * HDz + e];
      for (int d = 0; d < HDz; d++) acc += qb[t * HDz + d] * St[d * HDz + e];
      o[(rowbase + t) * DIMz + colbase + e] = f2bf(acc);
    }
    __syncthreads();

    // phase 5: S <- diag(g_tot) S + k_dec^T V
    for (int idx = tid; idx < HDz * HDz; idx += 256) {
      int d = idx >> 6, e = idx & 63;
      float acc = gtot[d] * St[idx];
      for (int t = 0; t < CHz; t++) acc += kd[t * HDz + d] * vs[t * HDz + e];
      St[idx] = acc;
    }
    __syncthreads();
  }
}

// ---------------------------------------------------------------- driver
extern "C" void kernel_launch(void* const* d_in, const int* in_sizes, int n_in,
                              void* d_out, int out_size, void* d_ws, size_t ws_size,
                              hipStream_t stream) {
  (void)in_sizes; (void)n_in; (void)out_size; (void)ws_size;

  const int*   tokens  = (const int*)  d_in[0];
  const float* embed   = (const float*)d_in[1];
  const float* norm1_w = (const float*)d_in[2];
  const float* norm2_w = (const float*)d_in[3];
  const float* Wq      = (const float*)d_in[4];
  const float* Wk      = (const float*)d_in[5];
  const float* Wv      = (const float*)d_in[6];
  const float* Wg      = (const float*)d_in[7];
  const float* Wo      = (const float*)d_in[8];
  const float* w1      = (const float*)d_in[9];
  const float* w3      = (const float*)d_in[10];   // dict order: w1, w3, w2
  const float* w2      = (const float*)d_in[11];
  const float* fnw     = (const float*)d_in[12];
  const float* head_w  = (const float*)d_in[13];
  const float* head_b  = (const float*)d_in[14];
  float* out = (float*)d_out;

  size_t off = 0;
  auto carve = [&](size_t bytes) -> void* {
    void* p = (char*)d_ws + off;
    off += (bytes + 255) & ~(size_t)255;
    return p;
  };
  // activations
  float* x   = (float*)carve((size_t)Mz * DIMz * 4);   // residual stream f32
  u16*   hbf = (u16*)  carve((size_t)Mz * DIMz * 2);   // normalized acts bf16
  float* qf  = (float*)carve((size_t)Mz * DIMz * 4);
  float* kf  = (float*)carve((size_t)Mz * DIMz * 4);
  float* vf  = (float*)carve((size_t)Mz * DIMz * 4);
  float* gf  = (float*)carve((size_t)Mz * DIMz * 4);
  u16*   obf = (u16*)  carve((size_t)Mz * DIMz * 2);   // attention out bf16
  float* ub  = (float*)carve((size_t)Mz * FFz * 4);
  float* v3b = (float*)carve((size_t)Mz * FFz * 4);
  u16*   tbf = (u16*)  carve((size_t)Mz * FFz * 2);
  // transposed bf16 weights (Wt[N][K])
  const size_t DD = (size_t)DIMz * DIMz, DF = (size_t)DIMz * FFz;
  u16* WqT = (u16*)carve((size_t)Lz * DD * 2);
  u16* WkT = (u16*)carve((size_t)Lz * DD * 2);
  u16* WvT = (u16*)carve((size_t)Lz * DD * 2);
  u16* WgT = (u16*)carve((size_t)Lz * DD * 2);
  u16* WoT = (u16*)carve((size_t)Lz * DD * 2);
  u16* w1T = (u16*)carve((size_t)Lz * DF * 2);
  u16* w3T = (u16*)carve((size_t)Lz * DF * 2);
  u16* w2T = (u16*)carve((size_t)Lz * DF * 2);
  u16* hwT = (u16*)carve((size_t)Vz * DIMz * 2);

  const dim3 blk(256);

  // ---- one-time (per launch) weight convert+transpose, f32 -> bf16 [N][K]
  {
    dim3 gDD(DIMz / 32, DIMz / 32);       // K=DIM, N=DIM
    dim3 gDF(FFz  / 32, DIMz / 32);       // K=DIM, N=FF
    dim3 gFD(DIMz / 32, FFz  / 32);       // K=FF,  N=DIM
    for (int l = 0; l < Lz; l++) {
      wtrans_kernel<<<gDD, blk, 0, stream>>>(Wq + l * DD, WqT + l * DD, DIMz, DIMz);
      wtrans_kernel<<<gDD, blk, 0, stream>>>(Wk + l * DD, WkT + l * DD, DIMz, DIMz);
      wtrans_kernel<<<gDD, blk, 0, stream>>>(Wv + l * DD, WvT + l * DD, DIMz, DIMz);
      wtrans_kernel<<<gDD, blk, 0, stream>>>(Wg + l * DD, WgT + l * DD, DIMz, DIMz);
      wtrans_kernel<<<gDD, blk, 0, stream>>>(Wo + l * DD, WoT + l * DD, DIMz, DIMz);
      wtrans_kernel<<<gDF, blk, 0, stream>>>(w1 + l * DF, w1T + l * DF, DIMz, FFz);
      wtrans_kernel<<<gDF, blk, 0, stream>>>(w3 + l * DF, w3T + l * DF, DIMz, FFz);
      wtrans_kernel<<<gFD, blk, 0, stream>>>(w2 + l * DF, w2T + l * DF, FFz, DIMz);
    }
    dim3 gHV(Vz / 32, DIMz / 32);         // K=DIM, N=V
    wtrans_kernel<<<gHV, blk, 0, stream>>>(head_w, hwT, DIMz, Vz);
  }

  const dim3 gProj(DIMz / 128, Mz / 128);   // N=1024
  const dim3 gFF  (FFz  / 128, Mz / 128);   // N=4096
  const dim3 gHead(Vz   / 128, Mz / 128);   // N=32000 (250 tiles, exact)

  embed_kernel<<<Mz, blk, 0, stream>>>(tokens, embed, x);

  for (int l = 0; l < Lz; l++) {
    rmsnorm_kernel<<<Mz, blk, 0, stream>>>(x, norm1_w + l * DIMz, hbf);

    gemm_kernel<0><<<gProj, blk, 0, stream>>>(hbf, WqT + l * DD, qf, nullptr, DIMz, DIMz);
    gemm_kernel<0><<<gProj, blk, 0, stream>>>(hbf, WkT + l * DD, kf, nullptr, DIMz, DIMz);
    gemm_kernel<0><<<gProj, blk, 0, stream>>>(hbf, WvT + l * DD, vf, nullptr, DIMz, DIMz);
    gemm_kernel<0><<<gProj, blk, 0, stream>>>(hbf, WgT + l * DD, gf, nullptr, DIMz, DIMz);

    gla_kernel<<<Bz * Hz, blk, 0, stream>>>(qf, kf, vf, gf, obf);

    gemm_kernel<1><<<gProj, blk, 0, stream>>>(obf, WoT + l * DD, x, nullptr, DIMz, DIMz);

    rmsnorm_kernel<<<Mz, blk, 0, stream>>>(x, norm2_w + l * DIMz, hbf);

    gemm_kernel<0><<<gFF, blk, 0, stream>>>(hbf, w1T + l * DF, ub,  nullptr, FFz, DIMz);
    gemm_kernel<0><<<gFF, blk, 0, stream>>>(hbf, w3T + l * DF, v3b, nullptr, FFz, DIMz);

    silumul_kernel<<<(Mz * FFz) / 256, blk, 0, stream>>>(ub, v3b, tbf, Mz * FFz);

    gemm_kernel<1><<<gProj, blk, 0, stream>>>(tbf, w2T + l * DF, x, nullptr, DIMz, FFz);
  }

  rmsnorm_kernel<<<Mz, blk, 0, stream>>>(x, fnw, hbf);
  gemm_kernel<2><<<gHead, blk, 0, stream>>>(hbf, hwT, out, head_b, Vz, DIMz);
}
